// QuantizedMatMul_18700287607012
// MI455X (gfx1250) — 8-bit matrix operand order repaired and checked by reading, compiled, not run on hardware
//
#include <hip/hip_runtime.h>

typedef __attribute__((ext_vector_type(16))) int   v16i;
typedef __attribute__((ext_vector_type(8)))  float v8f;

#define M_TOTAL 65536
#define N_TOTAL 4096
#define K_TOTAL 128
#define NT      4

__global__ __launch_bounds__(256) void quant_x_kernel(
    const float* __restrict__ x, const float* __restrict__ xs,
    unsigned char* __restrict__ qx) {
  const float s = xs[0];
  size_t base = ((size_t)blockIdx.x * blockDim.x + threadIdx.x) * 16;
  const float4* __restrict__ xin = (const float4*)(x + base);
  uint4 ov;
  unsigned* o = (unsigned*)&ov;
#pragma unroll
  for (int i = 0; i < 4; ++i) {
    float4 f = xin[i];
    int p = 0;
    p = __builtin_amdgcn_cvt_pk_fp8_f32(f.x * s, f.y * s, p, false);
    p = __builtin_amdgcn_cvt_pk_fp8_f32(f.z * s, f.w * s, p, true);
    o[i] = (unsigned)p;
  }
  *(uint4*)(qx + base) = ov;
}

__global__ __launch_bounds__(256) void quant_y_kernel(
    const float* __restrict__ y, const float* __restrict__ ys,
    unsigned char* __restrict__ qyT) {
  const float s = ys[0];
  int n = blockIdx.x * blockDim.x + threadIdx.x;
  unsigned* __restrict__ orow = (unsigned*)(qyT + (size_t)n * K_TOTAL);
#pragma unroll 4
  for (int k = 0; k < K_TOTAL; k += 4) {
    float f0 = y[(size_t)(k + 0) * N_TOTAL + n] * s;
    float f1 = y[(size_t)(k + 1) * N_TOTAL + n] * s;
    float f2 = y[(size_t)(k + 2) * N_TOTAL + n] * s;
    float f3 = y[(size_t)(k + 3) * N_TOTAL + n] * s;
    int p = 0;
    p = __builtin_amdgcn_cvt_pk_fp8_f32(f0, f1, p, false);
    p = __builtin_amdgcn_cvt_pk_fp8_f32(f2, f3, p, true);
    orow[k >> 2] = (unsigned)p;
  }
}

__global__ __launch_bounds__(256) void gemm_fp8_kernel(
    const unsigned char* __restrict__ qx,
    const unsigned char* __restrict__ qyT,
    const float* __restrict__ xs, const float* __restrict__ ys,
    float* __restrict__ out) {
  const float inv = 1.0f / (xs[0] * ys[0]);

  const int wave = blockIdx.x * (blockDim.x >> 5) + (threadIdx.x >> 5);
  const int lane = threadIdx.x & 31;
  const int half = lane >> 4;
  const int l15  = lane & 15;

  const int n_groups = N_TOTAL / (16 * NT);
  const int m0 = (wave / n_groups) * 16;
  const int n0 = (wave % n_groups) * (16 * NT);

  const unsigned char* __restrict__ arow = qx + (size_t)(m0 + l15) * K_TOTAL;
  v16i a;
#pragma unroll
  for (int b = 0; b < 2; ++b) {
#pragma unroll
    for (int j = 0; j < 4; ++j) {
      const int2 q = *(const int2*)(arow + b * 64 + j * 16 + half * 8);
      a[8 * b + 2 * j + 0] = q.x;
      a[8 * b + 2 * j + 1] = q.y;
    }
  }

#pragma unroll
  for (int nt = 0; nt < NT; ++nt) {
    const int n = n0 + nt * 16 + l15;
    const unsigned char* __restrict__ bcol = qyT + (size_t)n * K_TOTAL;
    v16i b;
#pragma unroll
    for (int g = 0; g < 8; ++g) {
      const int2 q = *(const int2*)(bcol + g * 16 + half * 8);
      b[2 * g + 0] = q.x;
      b[2 * g + 1] = q.y;
    }

    v8f c = {};
    c = __builtin_amdgcn_wmma_f32_16x16x128_fp8_fp8(a, b, (short)0, c,
                                                    false, false);

    float* __restrict__ orow = out + (size_t)(m0 + 8 * half) * N_TOTAL + n;
#pragma unroll
    for (int r = 0; r < 8; ++r) {
      orow[(size_t)r * N_TOTAL] = c[r] * inv;
    }
  }
}

extern "C" void kernel_launch(void* const* d_in, const int* in_sizes, int n_in,
                              void* d_out, int out_size, void* d_ws,
                              size_t ws_size, hipStream_t stream) {
  const float* x  = (const float*)d_in[0];
  const float* y  = (const float*)d_in[1];
  const float* xs = (const float*)d_in[2];
  const float* ys = (const float*)d_in[3];
  float* out = (float*)d_out;

  unsigned char* qx  = (unsigned char*)d_ws;
  unsigned char* qyT = qx + (size_t)M_TOTAL * K_TOTAL;

  {
    int threads = (M_TOTAL * K_TOTAL) / 16;
    quant_x_kernel<<<threads / 256, 256, 0, stream>>>(x, xs, qx);
  }
  quant_y_kernel<<<N_TOTAL / 256, 256, 0, stream>>>(y, ys, qyT);

  {
    int waves = (M_TOTAL / 16) * (N_TOTAL / (16 * NT));
    gemm_fp8_kernel<<<waves / 8, 256, 0, stream>>>(qx, qyT, xs, ys, out);
  }
}
